// CausalSelfAttention_36137854829197
// MI455X (gfx1250) — compile-verified, matrix operand order checked by reading, not run on hardware
//
#include <hip/hip_runtime.h>
#include <hip/hip_bf16.h>
#include <stdint.h>

#define BB 2
#define TT 2048
#define CC 2048
#define NH 16
#define NKV 4
#define HD 128
#define HALF 64

typedef __attribute__((ext_vector_type(16))) __bf16 v16bf;
typedef __attribute__((ext_vector_type(8)))  float  v8f;
typedef __attribute__((ext_vector_type(4)))  unsigned int v4u;
typedef __attribute__((ext_vector_type(4)))  int    v4i;
typedef __attribute__((ext_vector_type(8)))  int    v8i;

union FragU { v16bf v; uint4 u[2]; };

__device__ inline v8f wmma_bf16(v16bf a, v16bf b, v8f c) {
  return __builtin_amdgcn_wmma_f32_16x16x32_bf16(false, a, false, b, (short)0, c, false, false);
}

// Load a 16x32 bf16 A-fragment (or a 32x16 B-fragment from an [N][K]-stored
// matrix, global or LDS). ISA layout: lane holds row (lane&15); its 16
// elements are k0 + (lane>>4)*8 + {0..7} and the same +16.
__device__ inline v16bf load_frag(const __bf16* base, int ld,
                                  int row0, int k0, int lane) {
  const __bf16* p = base + (size_t)(row0 + (lane & 15)) * ld + k0 + ((lane >> 4) << 3);
  FragU f;
  f.u[0] = *(const uint4*)p;
  f.u[1] = *(const uint4*)(p + 16);
  return f.v;
}

// Tensor Data Mover: async-DMA a 2D bf16 tile (tile_d1 rows x tile_d0 elems,
// row stride stride_elems) from global into LDS, packed contiguously.
// D# per CDNA5 ISA. Tracked by TENSORcnt.
__device__ inline void tdm_load_2d(void* lds_dst, const void* gsrc,
                                   unsigned tile_d0, unsigned tile_d1,
                                   unsigned tensor_d0, unsigned tensor_d1,
                                   unsigned stride_elems) {
  unsigned long long ga = (unsigned long long)(uintptr_t)gsrc;
  unsigned lds = (unsigned)(uintptr_t)lds_dst;  // LDS byte offset
  v4u g0;
  g0.x = 1u;                                      // count=1, user descriptor
  g0.y = lds;                                     // lds_addr
  g0.z = (unsigned)(ga & 0xFFFFFFFFu);            // global_addr[31:0]
  g0.w = (unsigned)((ga >> 32) & 0x1FFFFFFu) | (2u << 30);  // [56:32] | type=2
  v8i g1;
  g1[0] = (int)0x00010000u;                                   // data_size=1 (2B)
  g1[1] = (int)((tensor_d0 & 0xFFFFu) << 16);                 // tensor_dim0[15:0]
  g1[2] = (int)(((tensor_d0 >> 16) & 0xFFFFu) |
                ((tensor_d1 & 0xFFFFu) << 16));               // tdim0[31:16]|tdim1[15:0]
  g1[3] = (int)(((tensor_d1 >> 16) & 0xFFFFu) |
                ((tile_d0 & 0xFFFFu) << 16));                 // tdim1[31:16]|tile_dim0
  g1[4] = (int)(tile_d1 & 0xFFFFu);                           // tile_dim1 (tile_dim2=0)
  g1[5] = (int)stride_elems;                                  // tensor_dim0_stride[31:0]
  g1[6] = 0;
  g1[7] = 0;
  v4i z4 = {0, 0, 0, 0};
#if __clang_major__ >= 23
  v8i z8 = {0, 0, 0, 0, 0, 0, 0, 0};
  __builtin_amdgcn_tensor_load_to_lds(g0, g1, z4, z4, z8, 0);
#else
  __builtin_amdgcn_tensor_load_to_lds(g0, g1, z4, z4, 0);
#endif
}

// Scalar wait: no-op for waves that issued no TDM ops (their TENSORcnt==0).
__device__ inline void tdm_wait() { __builtin_amdgcn_s_wait_tensorcnt(0); }

__global__ void k_f32_to_bf16(const float* __restrict__ in, __bf16* __restrict__ out, int n) {
  int i = blockIdx.x * blockDim.x + threadIdx.x;
  if (i < n) out[i] = (__bf16)in[i];
}

// W: [K][N] f32 row-major -> Wt: [N][K] bf16
__global__ void k_transpose_w(const float* __restrict__ W, __bf16* __restrict__ Wt, int N, int K) {
  int i = blockIdx.x * blockDim.x + threadIdx.x;
  if (i < N * K) {
    int n = i / K, k = i - n * K;
    Wt[i] = (__bf16)W[(size_t)k * N + n];
  }
}

// Fused projection: block = 4 waves = 64 rows x one full head (128 cols).
// Weight tile (128x32) staged in LDS by the TDM, double-buffered.
// mode 0: RMS-norm + RoPE epilogue -> bf16 [b][h][t][128]
// mode 1: plain epilogue -> bf16 transposed [b][h][128][t]   (for V)
__global__ void __launch_bounds__(128)
k_qkv(const __bf16* __restrict__ X,   // [B*T][CC] bf16
      const __bf16* __restrict__ Wt,  // [nheads*128][CC] bf16
      const float* __restrict__ cosp, const float* __restrict__ sinp,
      __bf16* __restrict__ out, int nheads, int mode) {
  __shared__ __align__(16) __bf16 wt[2][128][32];
  const int lane = threadIdx.x & 31;
  const int wv   = threadIdx.x >> 5;
  const int m0   = blockIdx.x * 64 + wv * 16;
  const int h    = blockIdx.y;
  const __bf16* wh = Wt + (size_t)(h * 128) * CC;

  if (wv == 0) tdm_load_2d(&wt[0][0][0], wh, 32, 128, CC, 128, CC);

  v8f c[8] = {};
  v16bf a = load_frag(X, CC, m0, 0, lane);  // A prefetched one k-step ahead
  for (int k0 = 0; k0 < CC; k0 += 32) {
    const int cur = (k0 >> 5) & 1;
    tdm_wait();
    __syncthreads();
    if (wv == 0 && k0 + 32 < CC)
      tdm_load_2d(&wt[cur ^ 1][0][0], wh + k0 + 32, 32, 128, CC - (k0 + 32), 128, CC);
    v16bf acur = a;
    if (k0 + 32 < CC) a = load_frag(X, CC, m0, k0 + 32, lane);
    v16bf bnext = load_frag(&wt[cur][0][0], 32, 0, 0, lane);
#pragma unroll
    for (int j = 0; j < 8; ++j) {
      v16bf bcur = bnext;
      if (j < 7) bnext = load_frag(&wt[cur][0][0], 32, (j + 1) * 16, 0, lane);
      c[j] = wmma_bf16(acur, bcur, c[j]);
    }
  }

  const int b     = m0 / TT;
  const int tbase = (m0 % TT) + ((lane >> 4) << 3);
  const int col   = lane & 15;

  if (mode == 0) {
    float rms[8];
#pragma unroll
    for (int i = 0; i < 8; ++i) {
      float ss = 0.f;
#pragma unroll
      for (int j = 0; j < 8; ++j) { float v = c[j][i]; ss += v * v; }
      ss += __shfl_xor(ss, 1); ss += __shfl_xor(ss, 2);
      ss += __shfl_xor(ss, 4); ss += __shfl_xor(ss, 8);
      rms[i] = rsqrtf(ss * (1.0f / 128.0f) + 1.1920929e-7f);
    }
    __bf16* oh = out + ((size_t)(b * nheads + h) * TT) * 128;
#pragma unroll
    for (int j = 0; j < 4; ++j) {
      int d = j * 16 + col;  // 0..63
#pragma unroll
      for (int i = 0; i < 8; ++i) {
        int t = tbase + i;
        float x1 = c[j][i]     * rms[i];
        float x2 = c[j + 4][i] * rms[i];
        float cv = cosp[t * HALF + d];
        float sv = sinp[t * HALF + d];
        oh[(size_t)t * 128 + d]      = (__bf16)(x1 * cv - x2 * sv);
        oh[(size_t)t * 128 + d + 64] = (__bf16)(x1 * sv + x2 * cv);
      }
    }
  } else {
    __bf16* oh = out + ((size_t)(b * nheads + h) * 128) * TT;
#pragma unroll
    for (int j = 0; j < 8; ++j) {
      int d = j * 16 + col;
#pragma unroll
      for (int i = 0; i < 8; ++i)
        oh[(size_t)d * TT + tbase + i] = (__bf16)c[j][i];
    }
  }
}

// Flash attention. Block = 4 waves = 64 q rows of one head; key tiles of 32
// staged in LDS by the TDM (K tile 32x128, V tile 128x32), double-buffered.
__global__ void __launch_bounds__(128)
k_attn(const __bf16* __restrict__ Q,   // [B][NH][T][128]
       const __bf16* __restrict__ K,   // [B][NKV][T][128]
       const __bf16* __restrict__ Vt,  // [B][NKV][128][T]
       __bf16* __restrict__ Y) {       // [B][T][C]
  __shared__ __align__(16) __bf16 kts[2][32][128];
  __shared__ __align__(16) __bf16 vts[2][128][32];
  __shared__ __align__(16) __bf16 pst[4][16][32];
  const int lane = threadIdx.x & 31;
  const int wv   = threadIdx.x >> 5;
  const int q0   = blockIdx.x * 64 + wv * 16;
  const int h    = blockIdx.y;
  const int b    = blockIdx.z;
  const int hkv  = h / (NH / NKV);
  const __bf16* qh = Q  + ((size_t)(b * NH  + h)   * TT) * 128;
  const __bf16* kh = K  + ((size_t)(b * NKV + hkv) * TT) * 128;
  const __bf16* vh = Vt + ((size_t)(b * NKV + hkv) * 128) * TT;

  v16bf qf[4];
#pragma unroll
  for (int kc = 0; kc < 4; ++kc) qf[kc] = load_frag(qh, 128, q0, kc * 32, lane);

  v8f acc[8] = {};
  float mrow[8], lrow[8];
#pragma unroll
  for (int i = 0; i < 8; ++i) { mrow[i] = -3.0e38f; lrow[i] = 0.f; }

  const float scale = 0.08838834764831845f;  // 1/sqrt(128)
  const int rbase = q0 + ((lane >> 4) << 3);
  const int col   = lane & 15;
  const int ktend = blockIdx.x * 64 + 63;    // block-uniform causal bound

  if (wv == 0) {
    tdm_load_2d(&kts[0][0][0], kh, 128, 32, 128, TT, 128);
    tdm_load_2d(&vts[0][0][0], vh, 32, 128, TT, 128, TT);
  }

  for (int kt = 0; kt <= ktend; kt += 32) {
    const int cur = (kt >> 5) & 1;
    tdm_wait();
    __syncthreads();
    if (wv == 0 && kt + 32 <= ktend) {
      tdm_load_2d(&kts[cur ^ 1][0][0], kh + (size_t)(kt + 32) * 128,
                  128, 32, 128, TT - (kt + 32), 128);
      tdm_load_2d(&vts[cur ^ 1][0][0], vh + kt + 32,
                  32, 128, TT - (kt + 32), 128, TT);
    }
    if (kt > q0 + 15) continue;  // wave-uniform: this wave is past its diagonal

    v8f s0 = {}, s1 = {};
    v16bf b0 = load_frag(&kts[cur][0][0], 128, 0,  0, lane);
    v16bf b1 = load_frag(&kts[cur][0][0], 128, 16, 0, lane);
#pragma unroll
    for (int kc = 0; kc < 4; ++kc) {
      v16bf c0 = b0, c1 = b1;
      if (kc < 3) {
        b0 = load_frag(&kts[cur][0][0], 128, 0,  (kc + 1) * 32, lane);
        b1 = load_frag(&kts[cur][0][0], 128, 16, (kc + 1) * 32, lane);
      }
      s0 = wmma_bf16(qf[kc], c0, s0);
      s1 = wmma_bf16(qf[kc], c1, s1);
    }
    float alpha[8];
#pragma unroll
    for (int i = 0; i < 8; ++i) {
      int r = rbase + i;
      float v0 = (kt + col      > r) ? -3.0e38f : s0[i] * scale;
      float v1 = (kt + 16 + col > r) ? -3.0e38f : s1[i] * scale;
      float mx = fmaxf(v0, v1);
      mx = fmaxf(mx, __shfl_xor(mx, 1));
      mx = fmaxf(mx, __shfl_xor(mx, 2));
      mx = fmaxf(mx, __shfl_xor(mx, 4));
      mx = fmaxf(mx, __shfl_xor(mx, 8));
      float mnew = fmaxf(mrow[i], mx);
      float a  = __expf(mrow[i] - mnew);
      float p0 = __expf(v0 - mnew);
      float p1 = __expf(v1 - mnew);
      float rs = p0 + p1;
      rs += __shfl_xor(rs, 1); rs += __shfl_xor(rs, 2);
      rs += __shfl_xor(rs, 4); rs += __shfl_xor(rs, 8);
      lrow[i] = lrow[i] * a + rs;
      mrow[i] = mnew;
      alpha[i] = a;
      int row = ((lane >> 4) << 3) + i;
      pst[wv][row][col]      = (__bf16)p0;
      pst[wv][row][col + 16] = (__bf16)p1;
    }
#pragma unroll
    for (int j = 0; j < 8; ++j)
#pragma unroll
      for (int i = 0; i < 8; ++i) acc[j][i] *= alpha[i];
    // Re-read P in A-fragment layout from LDS (C-layout -> A-layout permute)
    FragU pf;
    const __bf16* pp = &pst[wv][lane & 15][(lane >> 4) << 3];
    pf.u[0] = *(const uint4*)pp;
    pf.u[1] = *(const uint4*)(pp + 16);
    v16bf bv = load_frag(&vts[cur][0][0], 32, 0, 0, lane);
#pragma unroll
    for (int j = 0; j < 8; ++j) {
      v16bf bcur = bv;
      if (j < 7) bv = load_frag(&vts[cur][0][0], 32, (j + 1) * 16, 0, lane);
      acc[j] = wmma_bf16(pf.v, bcur, acc[j]);
    }
  }

  float inv[8];
#pragma unroll
  for (int i = 0; i < 8; ++i) inv[i] = 1.0f / lrow[i];
  __bf16* yb = Y + (size_t)b * TT * CC + h * 128;
#pragma unroll
  for (int j = 0; j < 8; ++j) {
    int d = j * 16 + col;
#pragma unroll
    for (int i = 0; i < 8; ++i) {
      int t = rbase + i;
      yb[(size_t)t * CC + d] = (__bf16)(acc[j][i] * inv[i]);
    }
  }
}

// C[M][N] f32 = A[M][K] bf16 @ Bt[N][K] bf16 (+ bias); B tile via TDM.
__global__ void __launch_bounds__(128)
k_gemm_bias(const __bf16* __restrict__ A, const __bf16* __restrict__ Bt,
            const float* __restrict__ bias, float* __restrict__ Cm,
            int M, int N, int K) {
  __shared__ __align__(16) __bf16 bts[2][64][32];
  const int lane = threadIdx.x & 31;
  const int wv   = threadIdx.x >> 5;
  const int m0   = blockIdx.y * 64 + wv * 16;
  const int n0   = blockIdx.x * 64;
  const __bf16* bh = Bt + (size_t)n0 * K;

  if (wv == 0) tdm_load_2d(&bts[0][0][0], bh, 32, 64, K, 64, K);

  v8f c[4] = {};
  v16bf a = load_frag(A, K, m0, 0, lane);  // A prefetched one k-step ahead
  for (int k0 = 0; k0 < K; k0 += 32) {
    const int cur = (k0 >> 5) & 1;
    tdm_wait();
    __syncthreads();
    if (wv == 0 && k0 + 32 < K)
      tdm_load_2d(&bts[cur ^ 1][0][0], bh + k0 + 32, 32, 64, K - (k0 + 32), 64, K);
    v16bf acur = a;
    if (k0 + 32 < K) a = load_frag(A, K, m0, k0 + 32, lane);
    v16bf bnext = load_frag(&bts[cur][0][0], 32, 0, 0, lane);
#pragma unroll
    for (int j = 0; j < 4; ++j) {
      v16bf bcur = bnext;
      if (j < 3) bnext = load_frag(&bts[cur][0][0], 32, (j + 1) * 16, 0, lane);
      c[j] = wmma_bf16(acur, bcur, c[j]);
    }
  }
  const int rb  = m0 + ((lane >> 4) << 3);
  const int col = lane & 15;
#pragma unroll
  for (int j = 0; j < 4; ++j) {
    int n = n0 + j * 16 + col;
    float bv = bias ? bias[n] : 0.f;
#pragma unroll
    for (int i = 0; i < 8; ++i)
      Cm[(size_t)(rb + i) * N + n] = c[j][i] + bv;
  }
}

extern "C" void kernel_launch(void* const* d_in, const int* in_sizes, int n_in,
                              void* d_out, int out_size, void* d_ws, size_t ws_size,
                              hipStream_t stream) {
  const float* x    = (const float*)d_in[0];
  const float* cosp = (const float*)d_in[1];
  const float* sinp = (const float*)d_in[2];
  const float* Wq   = (const float*)d_in[3];
  const float* Wk   = (const float*)d_in[4];
  const float* Wv   = (const float*)d_in[5];
  const float* Wo   = (const float*)d_in[6];
  const float* bo   = (const float*)d_in[7];
  float* out = (float*)d_out;

  char* ws = (char*)d_ws;
  size_t off = 0;
  auto alloc = [&](size_t bytes) -> char* {
    char* p = ws + off;
    off = (off + bytes + 255) & ~(size_t)255;
    return p;
  };
  const size_t M = (size_t)BB * TT;  // 4096
  __bf16* xb  = (__bf16*)alloc(M * CC * 2);
  __bf16* wqt = (__bf16*)alloc((size_t)CC * CC * 2);
  __bf16* wkt = (__bf16*)alloc((size_t)NKV * HD * CC * 2);
  __bf16* wvt = (__bf16*)alloc((size_t)NKV * HD * CC * 2);
  __bf16* wot = (__bf16*)alloc((size_t)CC * CC * 2);
  __bf16* ql  = (__bf16*)alloc(M * CC * 2);        // [B][NH][T][128]
  __bf16* kl  = (__bf16*)alloc(M * NKV * HD * 2);  // [B][NKV][T][128]
  __bf16* vt  = (__bf16*)alloc(M * NKV * HD * 2);  // [B][NKV][128][T]
  __bf16* y   = (__bf16*)alloc(M * CC * 2);        // [B][T][C]

  { int n = (int)(M * CC);
    k_f32_to_bf16<<<(n + 255) / 256, 256, 0, stream>>>(x, xb, n); }
  { int n = CC * CC;
    k_transpose_w<<<(n + 255) / 256, 256, 0, stream>>>(Wq, wqt, CC, CC); }
  { int n = NKV * HD * CC;
    k_transpose_w<<<(n + 255) / 256, 256, 0, stream>>>(Wk, wkt, NKV * HD, CC); }
  { int n = NKV * HD * CC;
    k_transpose_w<<<(n + 255) / 256, 256, 0, stream>>>(Wv, wvt, NKV * HD, CC); }
  { int n = CC * CC;
    k_transpose_w<<<(n + 255) / 256, 256, 0, stream>>>(Wo, wot, CC, CC); }

  dim3 blk(128);
  k_qkv<<<dim3((unsigned)(M / 64), NH),  blk, 0, stream>>>(xb, wqt, cosp, sinp, ql, NH, 0);
  k_qkv<<<dim3((unsigned)(M / 64), NKV), blk, 0, stream>>>(xb, wkt, cosp, sinp, kl, NKV, 0);
  k_qkv<<<dim3((unsigned)(M / 64), NKV), blk, 0, stream>>>(xb, wvt, cosp, sinp, vt, NKV, 1);
  k_attn<<<dim3(TT / 64, NH, BB), blk, 0, stream>>>(ql, kl, vt, y);
  k_gemm_bias<<<dim3(CC / 64, (unsigned)(M / 64)), blk, 0, stream>>>(y, wot, bo, out,
                                                                    (int)M, CC, CC);
}
